// CrossAttention_26551487824629
// MI455X (gfx1250) — compile-verified, matrix operand order checked by reading, not run on hardware
//
#include <hip/hip_runtime.h>
#include <hip/hip_bf16.h>
#include <stdint.h>

// ---------------------------------------------------------------------------
// CrossAttention forward for MI455X (gfx1250, wave32, WMMA).
//   B=4, N=2048, N2=1024, DIM=1024, H=16, HD=64, SCALE=0.125
// All matmuls via v_wmma_f32_16x16x32_bf16 (f32 accumulate).
// GEMM wave tile 64x64 (16 acc frags): 0.5 fragment-loads per WMMA.
// ---------------------------------------------------------------------------

typedef __attribute__((ext_vector_type(16))) __bf16 v16bf;
typedef __attribute__((ext_vector_type(8)))  __bf16 v8bf;
typedef __attribute__((ext_vector_type(8)))  float  v8f;

#define WMMA_BF16(a, b, c) \
  __builtin_amdgcn_wmma_f32_16x16x32_bf16(false, (a), false, (b), (short)0, (c), false, false)

__device__ __forceinline__ unsigned short f2bf(float f) {
  unsigned int u = __float_as_uint(f);
  u += 0x7FFFu + ((u >> 16) & 1u);      // round-to-nearest-even
  return (unsigned short)(u >> 16);
}

__device__ __forceinline__ v8f zero_v8f() {
  v8f z;
#pragma unroll
  for (int t = 0; t < 8; ++t) z[t] = 0.f;
  return z;
}

// Load one 16x32 bf16 WMMA fragment (A-layout; B-layout is the symmetric
// transpose so the same loader serves both when the source is K-contiguous).
// lane L: row = row0 + (L&15); K chunks at k0 + (L>>4)*8 and k0+16+(L>>4)*8.
__device__ __forceinline__ v16bf load_frag_g(const unsigned short* base,
                                             long long row0, int stride, int k0) {
  const int lane = threadIdx.x & 31;
  const unsigned short* p =
      base + (row0 + (lane & 15)) * (long long)stride + k0 + ((lane >> 4) << 3);
  v8bf lo = *(const v8bf*)p;
  v8bf hi = *(const v8bf*)(p + 16);
  v16bf r;
#pragma unroll
  for (int t = 0; t < 8; ++t) { r[t] = lo[t]; r[8 + t] = hi[t]; }
  return r;
}

// ---------------------------------------------------------------------------
// f32 -> bf16 elementwise convert
// ---------------------------------------------------------------------------
__global__ __launch_bounds__(256)
void cvt_bf16_kernel(const float* __restrict__ in, unsigned short* __restrict__ out, int n) {
  int i = blockIdx.x * 256 + threadIdx.x;
  if (i < n) out[i] = f2bf(in[i]);
}

// ---------------------------------------------------------------------------
// Transpose + convert weights: in[K][Nc] f32 -> out[Nc][K] bf16 (scaled).
// 32x32 LDS tile, fully coalesced both sides.
// ---------------------------------------------------------------------------
__global__ __launch_bounds__(256)
void transpose_cvt_kernel(const float* __restrict__ in, unsigned short* __restrict__ out,
                          int K, int Nc, float scale) {
  __shared__ float tile[32][33];
  const int tx = threadIdx.x, ty = threadIdx.y;    // 32 x 8
  const int bx = blockIdx.x * 32;                  // along Nc
  const int by = blockIdx.y * 32;                  // along K
#pragma unroll
  for (int t = 0; t < 4; ++t) {
    int r = ty + t * 8;
    tile[r][tx] = in[(long long)(by + r) * Nc + (bx + tx)];
  }
  __syncthreads();
#pragma unroll
  for (int t = 0; t < 4; ++t) {
    int rn = ty + t * 8;                           // Nc index within tile
    out[(long long)(bx + rn) * K + (by + tx)] = f2bf(tile[tx][rn] * scale);
  }
}

// ---------------------------------------------------------------------------
// bf16 WMMA GEMM: C[M,Nc] = A[M,K] * Bt[Nc,K]^T  (Bt is pre-transposed).
// Block = 256 thr = 8 waves (2x4), wave tile 64x64, block tile 128x256.
// mode 0: store bf16 row-major (Q projection)
// mode 1: KV split: cols<1024 -> Kb[b,h,n2,d]; cols>=1024 -> Vt[b,h,d,n2]
// mode 2: f32 + bias -> final output
// ---------------------------------------------------------------------------
__global__ __launch_bounds__(256)
void gemm_bf16_kernel(const unsigned short* __restrict__ A,
                      const unsigned short* __restrict__ Bt,
                      int M, int Nc, int K, int mode,
                      unsigned short* __restrict__ outU16,    // mode0: Qb ; mode1: Kb
                      unsigned short* __restrict__ outU16b,   // mode1: Vt
                      float* __restrict__ outF,               // mode2
                      const float* __restrict__ bias) {
  const int lane = threadIdx.x & 31;
  const int wave = threadIdx.x >> 5;
  const int wm = wave >> 2, wn = wave & 3;
  const int m0 = blockIdx.y * 128 + wm * 64;
  const int n0 = blockIdx.x * 256 + wn * 64;
  const int col0 = lane & 15, half = lane >> 4;

  v8f acc[4][4];
#pragma unroll
  for (int i = 0; i < 4; ++i)
#pragma unroll
    for (int j = 0; j < 4; ++j) acc[i][j] = zero_v8f();

  for (int k0 = 0; k0 < K; k0 += 32) {
    v16bf af[4], bf[4];
#pragma unroll
    for (int i = 0; i < 4; ++i) af[i] = load_frag_g(A, m0 + i * 16, K, k0);
#pragma unroll
    for (int j = 0; j < 4; ++j) bf[j] = load_frag_g(Bt, n0 + j * 16, K, k0);
#pragma unroll
    for (int i = 0; i < 4; ++i)
#pragma unroll
      for (int j = 0; j < 4; ++j)
        acc[i][j] = WMMA_BF16(af[i], bf[j], acc[i][j]);
  }

  // Epilogue. C layout: VGPR r -> row m0+tile+r+half*8, lanes hold 16 columns.
#pragma unroll
  for (int i = 0; i < 4; ++i) {
#pragma unroll
    for (int j = 0; j < 4; ++j) {
      const int c = n0 + j * 16 + col0;
#pragma unroll
      for (int r = 0; r < 8; ++r) {
        const int m = m0 + i * 16 + r + half * 8;
        const float v = acc[i][j][r];
        if (mode == 0) {
          outU16[(long long)m * Nc + c] = f2bf(v);
        } else if (mode == 1) {
          const int b = m >> 10, n2 = m & 1023;
          if (c < 1024) {
            const int h = c >> 6, d = c & 63;
            outU16[(((long long)(b * 16 + h) * 1024 + n2) << 6) + d] = f2bf(v);
          } else {
            const int cv = c - 1024, h = cv >> 6, d = cv & 63;
            outU16b[(((long long)(b * 16 + h) * 64 + d) << 10) + n2] = f2bf(v);
          }
        } else {
          outF[(long long)m * Nc + c] = v + bias[c];
        }
      }
    }
  }
}

// ---------------------------------------------------------------------------
// Flash attention: one wave per (b, h, 16-query tile). 32 keys per step.
// Qb  : [B*N, 1024] bf16 (pre-scaled by 1/sqrt(HD) via Wq)
// Kb  : [B,H,N2,64] bf16   Vt : [B,H,64,N2] bf16
// Ob  : [B*N, 1024] bf16 (cols h*64+d)
// ---------------------------------------------------------------------------
__global__ __launch_bounds__(128)
void attn_kernel(const unsigned short* __restrict__ Qb,
                 const unsigned short* __restrict__ Kb,
                 const unsigned short* __restrict__ Vt,
                 const int* __restrict__ mask,
                 unsigned short* __restrict__ Ob) {
  __shared__ __align__(16) unsigned short ldsP[4][16 * 32];   // per-wave P staging
  const int lane = threadIdx.x & 31;
  const int wave = threadIdx.x >> 5;
  const int task = blockIdx.x * 4 + wave;     // 8192 tasks
  const int qt = task & 127;                  // query tile (of 16 rows)
  const int head = task >> 7;                 // 0..63  (= b*16 + h)
  const int b = head >> 4, h = head & 15;
  const int col0 = lane & 15, half = lane >> 4;

  const unsigned short* kbase = Kb + (long long)head * (1024 * 64);
  const unsigned short* vbase = Vt + (long long)head * (64 * 1024);

  const long long qrow0 = (long long)b * 2048 + qt * 16;
  const v16bf q0 = load_frag_g(Qb, qrow0, 1024, h * 64);
  const v16bf q1 = load_frag_g(Qb, qrow0, 1024, h * 64 + 32);

  float mrow[8], lrow[8];
  v8f of[4];
#pragma unroll
  for (int r = 0; r < 8; ++r) { mrow[r] = -3.0e38f; lrow[r] = 0.f; }
#pragma unroll
  for (int g = 0; g < 4; ++g) of[g] = zero_v8f();

  for (int j0 = 0; j0 < 1024; j0 += 32) {
    if (j0 + 32 < 1024) {   // hint next K/V tiles -> global_prefetch_b8
      __builtin_prefetch(kbase + (long long)(j0 + 32 + col0) * 64, 0, 1);
      __builtin_prefetch(vbase + (long long)col0 * 1024 + j0 + 32, 0, 1);
    }
    // scores: S[16q x 32k] as two 16x16 frags, sum over d = two chained WMMAs
    const v16bf k00 = load_frag_g(kbase, j0,      64, 0);
    const v16bf k01 = load_frag_g(kbase, j0,      64, 32);
    const v16bf k10 = load_frag_g(kbase, j0 + 16, 64, 0);
    const v16bf k11 = load_frag_g(kbase, j0 + 16, 64, 32);
    v8f s0 = WMMA_BF16(q1, k01, zero_v8f());
    s0 = WMMA_BF16(q0, k00, s0);
    v8f s1 = WMMA_BF16(q1, k11, zero_v8f());
    s1 = WMMA_BF16(q0, k10, s1);

    const int keep0 = mask[b * 1024 + j0 + col0];
    const int keep1 = mask[b * 1024 + j0 + 16 + col0];

#pragma unroll
    for (int r = 0; r < 8; ++r) {
      const float a0 = keep0 ? s0[r] : -3.0e38f;
      const float a1 = keep1 ? s1[r] : -3.0e38f;
      float tmax = fmaxf(a0, a1);
      tmax = fmaxf(tmax, __shfl_xor(tmax, 1, 32));
      tmax = fmaxf(tmax, __shfl_xor(tmax, 2, 32));
      tmax = fmaxf(tmax, __shfl_xor(tmax, 4, 32));
      tmax = fmaxf(tmax, __shfl_xor(tmax, 8, 32));
      const float nm = fmaxf(mrow[r], tmax);
      const float alpha = __expf(mrow[r] - nm);
      const float p0 = keep0 ? __expf(s0[r] - nm) : 0.f;
      const float p1 = keep1 ? __expf(s1[r] - nm) : 0.f;
      float rs = p0 + p1;
      rs += __shfl_xor(rs, 1, 32);
      rs += __shfl_xor(rs, 2, 32);
      rs += __shfl_xor(rs, 4, 32);
      rs += __shfl_xor(rs, 8, 32);
      lrow[r] = lrow[r] * alpha + rs;
      mrow[r] = nm;
#pragma unroll
      for (int g = 0; g < 4; ++g) of[g][r] *= alpha;
      // stage P (C-layout) into LDS for relayout to A-fragment
      const int mr = r + half * 8;
      ldsP[wave][mr * 32 + col0]      = f2bf(p0);
      ldsP[wave][mr * 32 + 16 + col0] = f2bf(p1);
    }
    asm volatile("" ::: "memory");   // order LDS stores before cross-lane loads

    // P as 16x32 A-fragment (wave-internal; LDS in-order per wave)
    const unsigned short* pp = &ldsP[wave][col0 * 32 + half * 8];
    v8bf plo = *(const v8bf*)pp;
    v8bf phi = *(const v8bf*)(pp + 16);
    v16bf pf;
#pragma unroll
    for (int t = 0; t < 8; ++t) { pf[t] = plo[t]; pf[8 + t] = phi[t]; }

    // O += P * V : 4 d-groups of 16 columns
#pragma unroll
    for (int g = 0; g < 4; ++g) {
      const v16bf vf = load_frag_g(vbase, g * 16, 1024, j0);
      of[g] = WMMA_BF16(pf, vf, of[g]);
    }
    asm volatile("" ::: "memory");   // keep next iter's stores after these loads
  }

  // finalize: divide by row sums, write bf16 rows into Ob
#pragma unroll
  for (int r = 0; r < 8; ++r) {
    const float inv = lrow[r] > 0.f ? 1.0f / lrow[r] : 0.f;
    const int m = qt * 16 + r + half * 8;
    const long long orow = ((long long)b * 2048 + m) * 1024 + h * 64;
#pragma unroll
    for (int g = 0; g < 4; ++g)
      Ob[orow + g * 16 + col0] = f2bf(of[g][r] * inv);
  }
}

// ---------------------------------------------------------------------------
extern "C" void kernel_launch(void* const* d_in, const int* in_sizes, int n_in,
                              void* d_out, int out_size, void* d_ws, size_t ws_size,
                              hipStream_t stream) {
  (void)in_sizes; (void)n_in; (void)out_size; (void)ws_size;
  const float* x     = (const float*)d_in[0];   // [4,2048,1024]
  const float* y     = (const float*)d_in[1];   // [4,1024,1024]
  const int*   pad   = (const int*)d_in[2];     // [4,1024]
  const float* Wq    = (const float*)d_in[3];   // [1024,1024]
  const float* Wkv   = (const float*)d_in[4];   // [1024,2048]
  const float* Wproj = (const float*)d_in[5];   // [1024,1024]
  const float* bproj = (const float*)d_in[6];   // [1024]
  float* out = (float*)d_out;                   // [4,2048,1024]

  char* ws = (char*)d_ws;
  unsigned short* xb      = (unsigned short*)ws; ws += (size_t)8192 * 1024 * 2;  // 16 MB
  unsigned short* yb      = (unsigned short*)ws; ws += (size_t)4096 * 1024 * 2;  //  8 MB
  unsigned short* wq_t    = (unsigned short*)ws; ws += (size_t)1024 * 1024 * 2;  //  2 MB
  unsigned short* wkv_t   = (unsigned short*)ws; ws += (size_t)2048 * 1024 * 2;  //  4 MB
  unsigned short* wproj_t = (unsigned short*)ws; ws += (size_t)1024 * 1024 * 2;  //  2 MB
  unsigned short* qb      = (unsigned short*)ws; ws += (size_t)8192 * 1024 * 2;  // 16 MB
  unsigned short* kb      = (unsigned short*)ws; ws += (size_t)64 * 1024 * 64 * 2; // 8 MB
  unsigned short* vt      = (unsigned short*)ws; ws += (size_t)64 * 64 * 1024 * 2; // 8 MB
  unsigned short* ob      = (unsigned short*)ws;                                  // 16 MB

  const dim3 tb32x8(32, 8);

  // 1) precision conversion + weight transposes (SCALE folded into Wq)
  cvt_bf16_kernel<<<(8192 * 1024) / 256, 256, 0, stream>>>(x, xb, 8192 * 1024);
  cvt_bf16_kernel<<<(4096 * 1024) / 256, 256, 0, stream>>>(y, yb, 4096 * 1024);
  transpose_cvt_kernel<<<dim3(32, 32), tb32x8, 0, stream>>>(Wq,    wq_t,    1024, 1024, 0.125f);
  transpose_cvt_kernel<<<dim3(64, 32), tb32x8, 0, stream>>>(Wkv,   wkv_t,   1024, 2048, 1.0f);
  transpose_cvt_kernel<<<dim3(32, 32), tb32x8, 0, stream>>>(Wproj, wproj_t, 1024, 1024, 1.0f);

  // 2) Q = x*Wq (bf16, pre-scaled); KV = y*Wkv split into K and V^T layouts
  gemm_bf16_kernel<<<dim3(4, 64), 256, 0, stream>>>(xb, wq_t, 8192, 1024, 1024, 0,
                                                    qb, nullptr, nullptr, nullptr);
  gemm_bf16_kernel<<<dim3(8, 32), 256, 0, stream>>>(yb, wkv_t, 4096, 2048, 1024, 1,
                                                    kb, vt, nullptr, nullptr);

  // 3) flash attention (8192 wave-tasks, 4 waves per block)
  attn_kernel<<<2048, 128, 0, stream>>>(qb, kb, vt, pad, ob);

  // 4) out = O*Wproj + bproj (f32)
  gemm_bf16_kernel<<<dim3(4, 64), 256, 0, stream>>>(ob, wproj_t, 8192, 1024, 1024, 2,
                                                    nullptr, nullptr, out, bproj);
}
